// Gaussian_Kernel_62749472195001
// MI455X (gfx1250) — matrix operand order repaired and checked by reading, not compiled, not run on hardware
//
#include <hip/hip_runtime.h>

// ---------------------------------------------------------------------------
// Gaussian kernel:  out[n,k,p] = exp(2*<y_p, w_k> - ||y_p||^2 - ||w_k||^2)
// y_p = 2x2x2x16 unfold patch of x (d2 = 128), p over 31^3, k over 512.
// Store-bandwidth bound (244 MB out @ 23.3 TB/s ~ 10.5 us); GEMM (15.6 GFLOP)
// done with v_wmma_f32_16x16x32_bf16, f32 accumulation.
// ---------------------------------------------------------------------------

typedef __attribute__((ext_vector_type(16))) __bf16 v16bf;
typedef __attribute__((ext_vector_type(8)))  float  v8f;

#define P_TOT            29791   // 31*31*31
#define POS_TILE         64
#define BLOCKS_PER_BATCH 466     // ceil(29791/64)
#define D2               128     // K dimension (16 channels * 8 taps)
#define NBATCH           4
#define D1               512
#define KROWS            128     // d1 rows per block
#define LDS_STRIDE       (D2 + 8) // pad to avoid LDS bank conflicts

__global__ __launch_bounds__(256)
void gaussian_wmma_kernel(const float* __restrict__ x,
                          const float* __restrict__ w,
                          float* __restrict__ out) {
    __shared__ __bf16 wt[KROWS][LDS_STRIDE];     // staged w tile (bf16)
    __shared__ __bf16 yt[POS_TILE][LDS_STRIDE];  // staged patch tile (bf16)
    __shared__ float  ysq[POS_TILE];             // ||y_p||^2 (f32)
    __shared__ float  wsq[KROWS];                // ||w_k||^2 (f32)

    const int t     = threadIdx.x;
    const int nb    = blockIdx.x / BLOCKS_PER_BATCH;
    const int pbase = (blockIdx.x % BLOCKS_PER_BATCH) * POS_TILE;
    const int k0    = blockIdx.y * KROWS;

    if (t < POS_TILE) ysq[t] = 0.0f;
    __syncthreads();

    // ---- stage w tile: 128 rows x 128 f32 -> bf16 in LDS -------------------
    {
        const int krow = t >> 1;            // 0..127
        const int j0   = (t & 1) * 64;      // half-row
        const float* wrow = w + (size_t)(k0 + krow) * D2 + j0;
#pragma unroll
        for (int gch = 0; gch < 8; ++gch) {
            float4 f0 = ((const float4*)wrow)[gch * 2 + 0];
            float4 f1 = ((const float4*)wrow)[gch * 2 + 1];
            union { __bf16 h[8]; uint4 u; } pk;
            pk.h[0] = (__bf16)f0.x; pk.h[1] = (__bf16)f0.y;
            pk.h[2] = (__bf16)f0.z; pk.h[3] = (__bf16)f0.w;
            pk.h[4] = (__bf16)f1.x; pk.h[5] = (__bf16)f1.y;
            pk.h[6] = (__bf16)f1.z; pk.h[7] = (__bf16)f1.w;
            *(uint4*)&wt[krow][j0 + gch * 8] = pk.u;
        }
    }

    // ---- ||w_k||^2 in f32 (w rows are L2-resident) -------------------------
    if (t < KROWS) {
        const float4* wr = (const float4*)(w + (size_t)(k0 + t) * D2);
        float s = 0.0f;
#pragma unroll
        for (int i = 0; i < 32; ++i) {
            float4 v = wr[i];
            s += v.x * v.x + v.y * v.y + v.z * v.z + v.w * v.w;
        }
        wsq[t] = s;
    }

    // ---- stage patch tile: 64 positions x 128 values, plus ||y||^2 ---------
#pragma unroll
    for (int it = 0; it < 4; ++it) {
        const int idx = it * 256 + t;       // 0..1023
        const int pos = idx >> 4;           // 0..63
        const int c   = idx & 15;           // channel 0..15
        const int p   = pbase + pos;
        const int pc  = (p < P_TOT) ? p : (P_TOT - 1);
        const int d   = pc / 961;
        const int rm  = pc - d * 961;
        const int hh  = rm / 31;
        const int ww  = rm - hh * 31;
        const float* xb = x + ((((size_t)nb * 16 + c) * 32 + d) * 32 + hh) * 32 + ww;
        // 2x2x2 patch, feature order j = c*8 + kd*4 + kh*2 + kw
        float a0 = xb[0],    a1 = xb[1];
        float a2 = xb[32],   a3 = xb[33];
        float a4 = xb[1024], a5 = xb[1025];
        float a6 = xb[1056], a7 = xb[1057];
        union { __bf16 h[8]; uint4 u; } pk;
        pk.h[0] = (__bf16)a0; pk.h[1] = (__bf16)a1;
        pk.h[2] = (__bf16)a2; pk.h[3] = (__bf16)a3;
        pk.h[4] = (__bf16)a4; pk.h[5] = (__bf16)a5;
        pk.h[6] = (__bf16)a6; pk.h[7] = (__bf16)a7;
        *(uint4*)&yt[pos][c * 8] = pk.u;
        if (p < P_TOT) {
            float ss = a0 * a0 + a1 * a1 + a2 * a2 + a3 * a3 +
                       a4 * a4 + a5 * a5 + a6 * a6 + a7 * a7;
            atomicAdd(&ysq[pos], ss);       // ds_add_f32
        }
    }
    __syncthreads();

    // ---- WMMA GEMM: each wave owns 16 d1 rows, 4 position sub-tiles --------
    const int wave = t >> 5;
    const int lane = t & 31;
    const int m    = lane & 15;             // M row (A) == N col (B/C)
    const int g    = lane >> 4;             // half-wave group

    v8f acc[4] = {v8f{}, v8f{}, v8f{}, v8f{}};

    const __bf16* wrow = &wt[wave * 16 + m][0];

#pragma unroll
    for (int s = 0; s < 4; ++s) {           // K chunks of 32
        union { v16bf v; uint4 u[2]; } A;
        // A 16x32 bf16 layout: a[0..7] <- K=8g..8g+7 ; a[8..15] <- K=16+8g..+7
        A.u[0] = *(const uint4*)(wrow + s * 32 + 8 * g);
        A.u[1] = *(const uint4*)(wrow + s * 32 + 16 + 8 * g);
#pragma unroll
        for (int ps = 0; ps < 4; ++ps) {
            union { v16bf v; uint4 u[2]; } B;
            // B 32x16 bf16 layout: b[0..7] <- K=8g..8g+7 ; b[8..15] <- K=16+8g..+7 at column n=m
            const __bf16* yr = &yt[ps * 16 + m][s * 32 + 8 * g];
            B.u[0] = *(const uint4*)(yr);
            B.u[1] = *(const uint4*)(yr + 16);
            acc[ps] = __builtin_amdgcn_wmma_f32_16x16x32_bf16(
                false, A.v, false, B.v, (short)0, acc[ps], false, false);
        }
    }

    // ---- epilogue: exp(2*cross - ysq - wsq), p-contiguous stores -----------
    float wsqr[8];
#pragma unroll
    for (int r = 0; r < 8; ++r) wsqr[r] = wsq[wave * 16 + r + 8 * g];

#pragma unroll
    for (int ps = 0; ps < 4; ++ps) {
        const int p = pbase + ps * 16 + m;
        if (p >= P_TOT) continue;
        const float yq = ysq[ps * 16 + m];
        float* ob = out + ((size_t)nb * D1 + (k0 + wave * 16 + 8 * g)) * P_TOT + p;
#pragma unroll
        for (int r = 0; r < 8; ++r) {
            float val = __expf(2.0f * acc[ps][r] - yq - wsqr[r]);
            ob[(size_t)r * P_TOT] = val;
        }
    }
}

extern "C" void kernel_launch(void* const* d_in, const int* in_sizes, int n_in,
                              void* d_out, int out_size, void* d_ws, size_t ws_size,
                              hipStream_t stream) {
    (void)in_sizes; (void)n_in; (void)out_size; (void)d_ws; (void)ws_size;
    const float* x = (const float*)d_in[0];
    const float* w = (const float*)d_in[1];
    float* out = (float*)d_out;
    dim3 grid(NBATCH * BLOCKS_PER_BATCH, D1 / KROWS, 1);
    gaussian_wmma_kernel<<<grid, dim3(256), 0, stream>>>(x, w, out);
}
